// GNNModule_final_23948737643049
// MI455X (gfx1250) — matrix operand order repaired and checked by reading, not compiled, not run on hardware
//
#include <hip/hip_runtime.h>
#include <hip/hip_bf16.h>

// ---------------------------------------------------------------------------
// out = pd @ (y @ Wy^T) + x @ Wi^T + (bi + by)
// N=8192, M=16384, F=64.  Dominant cost: streaming pd (512 MB fp32) once.
// Strategy: 3xBF16-split WMMA (v_wmma_f32_16x16x32_bf16) -> HBM-bound ~22us.
// ---------------------------------------------------------------------------

typedef __attribute__((ext_vector_type(16))) __bf16 v16bf;
typedef __attribute__((ext_vector_type(8)))  float  v8f;
typedef __attribute__((ext_vector_type(8)))  float  v8flt;

#define N_DIM 8192
#define M_DIM 16384
#define F_DIM 64

// ---------------------------------------------------------------------------
// Kernel 1: z = y @ Wy^T  [M,64], stored transposed + split into bf16 hi/lo:
//   zT_hi[o*M + m], zT_lo[o*M + m]
// ---------------------------------------------------------------------------
__global__ __launch_bounds__(64)
void build_z_kernel(const float* __restrict__ y,
                    const float* __restrict__ Wy,
                    __bf16* __restrict__ zt_hi,
                    __bf16* __restrict__ zt_lo) {
    __shared__ float yrow[F_DIM];
    const int m = blockIdx.x;
    const int o = threadIdx.x;
    yrow[o] = y[(size_t)m * F_DIM + o];
    __syncthreads();
    const float* w = Wy + (size_t)o * F_DIM;
    float s = 0.0f;
#pragma unroll
    for (int f = 0; f < F_DIM; ++f) s += yrow[f] * w[f];
    __bf16 hi = (__bf16)s;
    __bf16 lo = (__bf16)(s - (float)hi);
    zt_hi[(size_t)o * M_DIM + m] = hi;
    zt_lo[(size_t)o * M_DIM + m] = lo;
}

// ---------------------------------------------------------------------------
// Kernel 2: e = x @ Wi^T + bi + by   [N,64] fp32
// ---------------------------------------------------------------------------
__global__ __launch_bounds__(64)
void build_e_kernel(const float* __restrict__ x,
                    const float* __restrict__ Wi,
                    const float* __restrict__ bi,
                    const float* __restrict__ by,
                    float* __restrict__ e) {
    __shared__ float xrow[F_DIM];
    const int n = blockIdx.x;
    const int o = threadIdx.x;
    xrow[o] = x[(size_t)n * F_DIM + o];
    __syncthreads();
    const float* w = Wi + (size_t)o * F_DIM;
    float s = bi[o] + by[o];
#pragma unroll
    for (int f = 0; f < F_DIM; ++f) s += xrow[f] * w[f];
    e[(size_t)n * F_DIM + o] = s;
}

// ---------------------------------------------------------------------------
// Kernel 3: out = pd @ z + e  via 3xBF16-split WMMA.
// 512 blocks x 256 threads (8 wave32). Block = 16-row tile of pd.
// Wave w handles K in [w*2048, (w+1)*2048); partials reduced through LDS.
// ---------------------------------------------------------------------------
__global__ __launch_bounds__(256)
void gemm_pd_kernel(const float*  __restrict__ pd,
                    const __bf16* __restrict__ zt_hi,
                    const __bf16* __restrict__ zt_lo,
                    const float*  __restrict__ e,
                    float*        __restrict__ out) {
    __shared__ float red[8][16][F_DIM];   // 32 KB

    const int lane = threadIdx.x & 31;
    const int wave = threadIdx.x >> 5;
    const int hh   = lane >> 4;           // 0: lanes 0-15, 1: lanes 16-31
    const int l15  = lane & 15;
    const int tr   = blockIdx.x * 16;     // tile row base

    // A operand: this lane's pd row; K window offset matching B layout:
    // lanes 0-15 need K {0..15}, lanes 16-31 need K {16..31}.
    const float* prow = pd + (size_t)(tr + l15) * M_DIM;

    const int kstart = wave * 2048;
    const int kend   = kstart + 2048;

    v8f acc[4] = {{}, {}, {}, {}};

    for (int kc = kstart; kc < kend; kc += 32) {
        // Prefetch pd stream ~2KB ahead (global_prefetch_b8).
        if (kc + 512 < kend) __builtin_prefetch(prow + kc + 512, 0, 1);

        const int g0 = kc + hh * 16;
        v8flt f0 = *(const v8flt*)(prow + g0);        // elements 0..7
        v8flt f1 = *(const v8flt*)(prow + g0 + 8);    // elements 8..15

        v16bf a_hi, a_lo;
#pragma unroll
        for (int i = 0; i < 8; ++i) {
            float v = f0[i];
            __bf16 h = (__bf16)v;
            a_hi[i] = h;
            a_lo[i] = (__bf16)(v - (float)h);
        }
#pragma unroll
        for (int i = 0; i < 8; ++i) {
            float v = f1[i];
            __bf16 h = (__bf16)v;
            a_hi[8 + i] = h;
            a_lo[8 + i] = (__bf16)(v - (float)h);
        }

        // B operand: contiguous k order, same as A: lanes 0-15 hold K=kc..kc+15,
        // lanes 16-31 hold K=kc+16..kc+31, column = c*16 + (lane&15).
        const int koff = kc + (hh << 4);
#pragma unroll
        for (int c = 0; c < 4; ++c) {
            const size_t boff = (size_t)(c * 16 + l15) * M_DIM + koff;
            v16bf b_hi = *(const v16bf*)(zt_hi + boff);
            v16bf b_lo = *(const v16bf*)(zt_lo + boff);
            acc[c] = __builtin_amdgcn_wmma_f32_16x16x32_bf16(
                false, a_hi, false, b_hi, (short)0, acc[c], false, false);
            acc[c] = __builtin_amdgcn_wmma_f32_16x16x32_bf16(
                false, a_hi, false, b_lo, (short)0, acc[c], false, false);
            acc[c] = __builtin_amdgcn_wmma_f32_16x16x32_bf16(
                false, a_lo, false, b_hi, (short)0, acc[c], false, false);
        }
    }

    // C/D layout: VGPR v -> row (v + 8*hh), col = c*16 + (lane&15).
#pragma unroll
    for (int c = 0; c < 4; ++c) {
#pragma unroll
        for (int v = 0; v < 8; ++v) {
            red[wave][hh * 8 + v][c * 16 + l15] = acc[c][v];
        }
    }
    __syncthreads();

    // 16x64 = 1024 elements, 256 threads -> 4 each: sum 8 partials + e.
#pragma unroll
    for (int j = 0; j < 4; ++j) {
        const int elem = threadIdx.x * 4 + j;
        const int rl = elem >> 6;
        const int cl = elem & 63;
        float s = 0.0f;
#pragma unroll
        for (int w = 0; w < 8; ++w) s += red[w][rl][cl];
        const size_t oi = (size_t)(tr + rl) * F_DIM + cl;
        out[oi] = s + e[oi];
    }
}

// ---------------------------------------------------------------------------
// Launch
// ---------------------------------------------------------------------------
extern "C" void kernel_launch(void* const* d_in, const int* in_sizes, int n_in,
                              void* d_out, int out_size, void* d_ws, size_t ws_size,
                              hipStream_t stream) {
    // setup_inputs order: x, y, deg_g, g_a1, pm, pd, theta_i_w, theta_i_b,
    //                     theta_y_w, theta_y_b
    const float* x  = (const float*)d_in[0];
    const float* y  = (const float*)d_in[1];
    const float* pd = (const float*)d_in[5];
    const float* Wi = (const float*)d_in[6];
    const float* bi = (const float*)d_in[7];
    const float* Wy = (const float*)d_in[8];
    const float* by = (const float*)d_in[9];
    float* out = (float*)d_out;

    char* ws = (char*)d_ws;
    const size_t zbytes = (size_t)F_DIM * M_DIM * sizeof(__bf16);  // 2 MB
    __bf16* zt_hi = (__bf16*)(ws);
    __bf16* zt_lo = (__bf16*)(ws + zbytes);
    float*  e     = (float*)(ws + 2 * zbytes);                     // 2 MB

    build_z_kernel<<<dim3(M_DIM), dim3(64), 0, stream>>>(y, Wy, zt_hi, zt_lo);
    build_e_kernel<<<dim3(N_DIM), dim3(64), 0, stream>>>(x, Wi, bi, by, e);
    gemm_pd_kernel<<<dim3(N_DIM / 16), dim3(256), 0, stream>>>(pd, zt_hi, zt_lo, e, out);
}
